// GNNEncoder_88313117540961
// MI455X (gfx1250) — compile-verified, matrix operand order checked by reading, not run on hardware
//
#include <hip/hip_runtime.h>
#include <hip/hip_bf16.h>
#include <math.h>

// ---------------------------------------------------------------------------
// GNN encoder: 3 x [GCNConv(WMMA GEMM + L2-resident scatter) + LN + SiLU],
// then global mean pool.  Targets MI455X (gfx1250, wave32, WMMA).
// ---------------------------------------------------------------------------

typedef __attribute__((ext_vector_type(16))) __bf16 v16bf;
typedef __attribute__((ext_vector_type(8)))  float  v8f;

#define D 128          // feature dim (fixed by reference)
#define LN_EPS 1e-5f

// ---------------------------------------------------------------------------
// generic float zeroing (float4 granularity; n4 = number of float4's)
__global__ void zero_f4(float4* p, int n4) {
    int i = blockIdx.x * blockDim.x + threadIdx.x;
    if (i < n4) p[i] = make_float4(0.f, 0.f, 0.f, 0.f);
}

// ---------------------------------------------------------------------------
// degree: deg[n] = 1 (self loop) + indegree over dst; then dinv = rsqrt(deg)
__global__ void deg_init(float* deg, int N) {
    int i = blockIdx.x * blockDim.x + threadIdx.x;
    if (i < N) deg[i] = 1.0f;
}
__global__ void deg_count(const int* __restrict__ dst, float* deg, int E) {
    int e = blockIdx.x * blockDim.x + threadIdx.x;
    if (e < E) atomicAdd(&deg[dst[e]], 1.0f);
}
__global__ void deg_rsqrt(float* deg, int N) {
    int i = blockIdx.x * blockDim.x + threadIdx.x;
    if (i < N) deg[i] = rsqrtf(deg[i]);   // deg >= 1 always
}

// ---------------------------------------------------------------------------
// Pack W (DxD f32, row-major W[k][n]) into the WMMA bf16 B-operand lane
// layout for V_WMMA_F32_16X16X32_BF16 (ISA 7.12.2, 16-bit 16x32 striping):
//   lane L<16 : col n = n0+L, elems 0..7 -> K = kb..kb+7, 8..15 -> kb+16..kb+23
//   lane L>=16: same col, kb offset +8
// Index: [kk(4)][ntile(8)][lane(32)][elem(16)]  -> one v16bf per lane per step.
__global__ void pack_w(const float* __restrict__ W, __bf16* __restrict__ Wp) {
    int t = blockIdx.x * blockDim.x + threadIdx.x;     // 1024 threads
    if (t >= 4 * 8 * 32) return;
    int lane = t & 31;
    int nt   = (t >> 5) & 7;
    int kk   = t >> 8;
    int n  = nt * 16 + (lane & 15);
    int kb = kk * 32 + ((lane & 16) ? 8 : 0);
    __bf16* dstp = Wp + (size_t)t * 16;
#pragma unroll
    for (int i = 0; i < 8; ++i) dstp[i]     = (__bf16)W[(kb + i) * D + n];
#pragma unroll
    for (int i = 0; i < 8; ++i) dstp[8 + i] = (__bf16)W[(kb + 16 + i) * D + n];
}

// ---------------------------------------------------------------------------
// H = X @ W  via v_wmma_f32_16x16x32_bf16.
// Block = 256 threads = 8 waves; wave w owns the 16-col tile n0 = 16*w,
// block owns the 16-row tile m0 = 16*blockIdx.x.  K = 128 -> 4 WMMA steps.
__global__ void gemm_wmma(const float* __restrict__ X,
                          const __bf16* __restrict__ Wp,
                          float* __restrict__ H, int N) {
    int lane = threadIdx.x & 31;
    int wave = threadIdx.x >> 5;
    int m0 = blockIdx.x * 16;
    if (m0 >= N) return;
    bool hi = lane >= 16;
    int row = m0 + (lane & 15);
    if (row >= N) row = N - 1;                 // clamp loads (stores guarded)

    v8f acc = {};
#pragma unroll
    for (int kk = 0; kk < 4; ++kk) {
        // A operand: f32 -> bf16 on the fly, ISA 16x32 16-bit striping
        int c0 = kk * 32 + (hi ? 8 : 0);
        const float* xr = X + (size_t)row * D + c0;
        v16bf a;
#pragma unroll
        for (int i = 0; i < 8; ++i) a[i]     = (__bf16)xr[i];
#pragma unroll
        for (int i = 0; i < 8; ++i) a[8 + i] = (__bf16)xr[16 + i];
        // B operand: pre-swizzled, one contiguous 32B load per lane
        v16bf b = *(const v16bf*)(Wp + ((size_t)((kk * 8 + wave) * 32 + lane) * 16));
        acc = __builtin_amdgcn_wmma_f32_16x16x32_bf16(
            /*neg_a=*/false, a, /*neg_b=*/false, b,
            /*c_mod=*/(short)0, acc, /*reuse_a=*/false, /*reuse_b=*/false);
    }
    // C/D layout: VGPR r -> M = m0 + r + (hi?8:0), Nc = n0 + (lane&15)
    int col   = wave * 16 + (lane & 15);
    int rbase = m0 + (hi ? 8 : 0);
#pragma unroll
    for (int r = 0; r < 8; ++r) {
        int rr = rbase + r;
        if (rr < N) H[(size_t)rr * D + col] = acc[r];
    }
}

// ---------------------------------------------------------------------------
// Edge scatter: AGG[dst] += H[src] * dinv[src]*dinv[dst].
// One 32-lane group per edge, 4 floats per lane.  AGG/H are L2-resident
// (25.6 MB each << 192 MB L2) so the f32 atomics resolve in L2.
__global__ void scatter_edges(const int* __restrict__ src,
                              const int* __restrict__ dst,
                              const float* __restrict__ dinv,
                              const float* __restrict__ H,
                              float* __restrict__ AGG, int E) {
    int t = blockIdx.x * blockDim.x + threadIdx.x;
    int e = t >> 5;
    if (e >= E) return;
    int l = t & 31;
    int s = src[e], d = dst[e];
    float w = dinv[s] * dinv[d];
    float4 hv = *(const float4*)(H + (size_t)s * D + l * 4);
    float* out = AGG + (size_t)d * D + l * 4;
    atomicAdd(out + 0, hv.x * w);
    atomicAdd(out + 1, hv.y * w);
    atomicAdd(out + 2, hv.z * w);
    atomicAdd(out + 3, hv.w * w);
}

// ---------------------------------------------------------------------------
// Fused: self-loop + bias + LayerNorm + residual + SiLU.
// One wave32 per node row, 4 feats/lane, cross-lane reduce via shfl_xor.
__global__ void fused_ln_silu(const float* __restrict__ AGG,
                              const float* __restrict__ H,
                              const float* __restrict__ dinv,
                              const float* __restrict__ Xin,
                              float* __restrict__ Xout,
                              const float* __restrict__ bias,
                              const float* __restrict__ gam,
                              const float* __restrict__ bet, int N) {
    int node = blockIdx.x * 8 + (threadIdx.x >> 5);
    if (node >= N) return;
    int l = threadIdx.x & 31;
    float di = dinv[node];
    float sw = di * di;                         // self-loop norm
    size_t base = (size_t)node * D + l * 4;

    float4 a  = *(const float4*)(AGG + base);
    float4 h  = *(const float4*)(H + base);
    float4 bb = *(const float4*)(bias + l * 4);
    float v0 = a.x + h.x * sw + bb.x;
    float v1 = a.y + h.y * sw + bb.y;
    float v2 = a.z + h.z * sw + bb.z;
    float v3 = a.w + h.w * sw + bb.w;

    float s  = v0 + v1 + v2 + v3;
    float sq = v0 * v0 + v1 * v1 + v2 * v2 + v3 * v3;
#pragma unroll
    for (int off = 16; off; off >>= 1) {
        s  += __shfl_xor(s,  off, 32);
        sq += __shfl_xor(sq, off, 32);
    }
    float mu  = s * (1.f / D);
    float var = sq * (1.f / D) - mu * mu;
    float inv = rsqrtf(var + LN_EPS);

    float4 gg = *(const float4*)(gam + l * 4);
    float4 be = *(const float4*)(bet + l * 4);
    float4 xv = *(const float4*)(Xin + base);

    float o0 = (v0 - mu) * inv * gg.x + be.x + xv.x;
    float o1 = (v1 - mu) * inv * gg.y + be.y + xv.y;
    float o2 = (v2 - mu) * inv * gg.z + be.z + xv.z;
    float o3 = (v3 - mu) * inv * gg.w + be.w + xv.w;
    float4 o;
    o.x = o0 / (1.f + __expf(-o0));
    o.y = o1 / (1.f + __expf(-o1));
    o.z = o2 / (1.f + __expf(-o2));
    o.w = o3 / (1.f + __expf(-o3));
    *(float4*)(Xout + base) = o;
}

// ---------------------------------------------------------------------------
// Global mean pool
__global__ void pool_accum(const float* __restrict__ X,
                           const int* __restrict__ batch,
                           float* __restrict__ sums,
                           float* __restrict__ counts, int N) {
    int t = blockIdx.x * blockDim.x + threadIdx.x;
    int node = t >> 5;
    if (node >= N) return;
    int l = t & 31;
    int g = batch[node];
    float4 xv = *(const float4*)(X + (size_t)node * D + l * 4);
    float* out = sums + (size_t)g * D + l * 4;
    atomicAdd(out + 0, xv.x);
    atomicAdd(out + 1, xv.y);
    atomicAdd(out + 2, xv.z);
    atomicAdd(out + 3, xv.w);
    if (l == 0) atomicAdd(&counts[g], 1.0f);
}
__global__ void pool_final(const float* __restrict__ sums,
                           const float* __restrict__ counts,
                           float* __restrict__ out, int GD) {
    int t = blockIdx.x * blockDim.x + threadIdx.x;
    if (t >= GD) return;
    float c = fmaxf(counts[t >> 7], 1.0f);
    out[t] = sums[t] / c;
}

// ---------------------------------------------------------------------------
extern "C" void kernel_launch(void* const* d_in, const int* in_sizes, int n_in,
                              void* d_out, int out_size, void* d_ws, size_t ws_size,
                              hipStream_t stream) {
    const float* x     = (const float*)d_in[0];
    const int*   ei    = (const int*)d_in[1];
    const int*   batch = (const int*)d_in[2];
    const float* W[3]    = { (const float*)d_in[3],  (const float*)d_in[7],  (const float*)d_in[11] };
    const float* bias[3] = { (const float*)d_in[4],  (const float*)d_in[8],  (const float*)d_in[12] };
    const float* gam[3]  = { (const float*)d_in[5],  (const float*)d_in[9],  (const float*)d_in[13] };
    const float* bet[3]  = { (const float*)d_in[6],  (const float*)d_in[10], (const float*)d_in[14] };

    const int N = in_sizes[2];          // batch id per node
    const int E = in_sizes[1] / 2;
    const int G = out_size / D;
    const int* srcI = ei;
    const int* dstI = ei + E;

    // workspace layout (256B aligned slabs)
    char* ws = (char*)d_ws;
    size_t off = 0;
    auto alloc = [&](size_t bytes) { char* p = ws + off; off += (bytes + 255) & ~(size_t)255; return p; };
    float*  dinv   = (float*)alloc((size_t)N * sizeof(float));
    float*  Hbuf   = (float*)alloc((size_t)N * D * sizeof(float));
    float*  AGG    = (float*)alloc((size_t)N * D * sizeof(float));
    float*  Xbuf   = (float*)alloc((size_t)N * D * sizeof(float));
    __bf16* Wp     = (__bf16*)alloc((size_t)3 * D * D * sizeof(__bf16));
    float*  sums   = (float*)alloc((size_t)G * D * sizeof(float));
    float*  counts = (float*)alloc((size_t)G * sizeof(float));
    (void)ws_size; (void)n_in;

    const int BT = 256;

    // ---- degrees / symmetric normalization
    deg_init <<<(N + BT - 1) / BT, BT, 0, stream>>>(dinv, N);
    deg_count<<<(E + BT - 1) / BT, BT, 0, stream>>>(dstI, dinv, E);
    deg_rsqrt<<<(N + BT - 1) / BT, BT, 0, stream>>>(dinv, N);

    // ---- pre-swizzle all three weight matrices into WMMA B-operand layout
    for (int l = 0; l < 3; ++l)
        pack_w<<<4, BT, 0, stream>>>(W[l], Wp + (size_t)l * D * D);

    // ---- zero pool accumulators (sums + counts are adjacent slabs; zero both)
    {
        int n4 = (G * D) / 4;
        zero_f4<<<(n4 + BT - 1) / BT, BT, 0, stream>>>((float4*)sums, n4);
        int c4 = (G + 3) / 4;
        zero_f4<<<(c4 + BT - 1) / BT, BT, 0, stream>>>((float4*)counts, c4);
    }

    // ---- three GCN blocks
    const float* xin = x;
    for (int l = 0; l < 3; ++l) {
        gemm_wmma<<<(N + 15) / 16, BT, 0, stream>>>(xin, Wp + (size_t)l * D * D, Hbuf, N);
        int n4 = (N * D) / 4;
        zero_f4<<<(n4 + BT - 1) / BT, BT, 0, stream>>>((float4*)AGG, n4);
        scatter_edges<<<((size_t)E * 32 + BT - 1) / BT, BT, 0, stream>>>(srcI, dstI, dinv, Hbuf, AGG, E);
        fused_ln_silu<<<(N + 7) / 8, BT, 0, stream>>>(AGG, Hbuf, dinv, xin, Xbuf,
                                                      bias[l], gam[l], bet[l], N);
        xin = Xbuf;   // layers 1,2 read+write Xbuf in place (per-thread element-wise: safe)
    }

    // ---- global mean pool
    pool_accum<<<((size_t)N * 32 + BT - 1) / BT, BT, 0, stream>>>(Xbuf, batch, sums, counts, N);
    pool_final<<<(G * D + BT - 1) / BT, BT, 0, stream>>>(sums, counts, (float*)d_out, G * D);
}
